// Attention1D_6038724018458
// MI455X (gfx1250) — matrix operand order repaired and checked by reading, not compiled, not run on hardware
//
#include <hip/hip_runtime.h>
#include <hip/hip_bf16.h>
#include <stdint.h>

// ---------------- problem constants ----------------
#define BATCH    8
#define CDIM     512      // channels (layernorm dim / model dim)
#define NSEQ     2048     // sequence length
#define HEADS    8
#define DHEAD    64
#define QKV_ROWS 1536     // 3 * HEADS*DHEAD
#define EPS      1e-5f

typedef __attribute__((ext_vector_type(16))) __bf16 v16bf;
typedef __attribute__((ext_vector_type(8)))  float  v8f;

// ---------------- helpers ----------------
__device__ __forceinline__ unsigned short f32_to_bf16(float f) {
  unsigned int u = __float_as_uint(f);
  unsigned int r = (u + 0x7FFFu + ((u >> 16) & 1u)) >> 16;   // round-nearest-even
  return (unsigned short)r;
}

__device__ __forceinline__ v8f zero8() {
  v8f z = {0.f, 0.f, 0.f, 0.f, 0.f, 0.f, 0.f, 0.f};
  return z;
}

struct FragAB { union { v16bf v; uint4 u4[2]; }; };

// A fragment: 16x32 bf16, source row-major [M][K] with leading dim ld (elements).
// lane L: M = L%16 ; element e: K = (e/8)*16 + (L/16)*8 + e%8  -> two b128 loads.
__device__ __forceinline__ FragAB load_frag_a(const unsigned short* base, int ld, int lane) {
  int row = lane & 15, half = lane >> 4;
  const unsigned short* p = base + (size_t)row * ld + half * 8;
  FragAB f;
  f.u4[0] = *(const uint4*)(p);
  f.u4[1] = *(const uint4*)(p + 16);
  return f;
}

// B fragment: 32x16 bf16 (KxN), source stored N-major with K contiguous: [N][K], ld.
// lane L: N = L%16 ; element e: K = (e/8)*16 + (L/16)*8 + e%8  -> two b128 loads.
__device__ __forceinline__ FragAB load_frag_b(const unsigned short* base, int ld, int lane) {
  int col = lane & 15, half = lane >> 4;
  const unsigned short* p = base + (size_t)col * ld + half * 8;
  FragAB f;
  f.u4[0] = *(const uint4*)(p);
  f.u4[1] = *(const uint4*)(p + 16);
  return f;
}

__device__ __forceinline__ v8f wmma_bf16(const FragAB& a, const FragAB& b, v8f c) {
  return __builtin_amdgcn_wmma_f32_16x16x32_bf16(false, a.v, false, b.v,
                                                 (short)0, c, false, false);
}

// CDNA5 async global->LDS copy (ASYNCcnt path). LDS byte offset is the low 32
// bits of the generic pointer (flat LDS addr = aperture || offset, ISA 10.2).
__device__ __forceinline__ void async_g2l_b128(void* lds_ptr, const void* gptr) {
  unsigned lds = (unsigned)(uintptr_t)lds_ptr;
  unsigned long long g = (unsigned long long)gptr;
  asm volatile("global_load_async_to_lds_b128 %0, %1, off"
               :: "v"(lds), "v"(g) : "memory");
}
__device__ __forceinline__ void wait_asynccnt0() {
  asm volatile("s_wait_asynccnt 0" ::: "memory");
}

// ---------------- kernel 0: f32 -> bf16 weight conversion ----------------
__global__ __launch_bounds__(256) void cvt_bf16_kernel(const float* __restrict__ src,
                                                       unsigned short* __restrict__ dst,
                                                       int n) {
  int i = blockIdx.x * 256 + threadIdx.x;
  if (i < n) dst[i] = f32_to_bf16(src[i]);
}

// ---------------- kernel 1: channel LayerNorm, write xn^T (b, n, c) bf16 ------
__global__ __launch_bounds__(256) void layernorm_kernel(const float* __restrict__ x,
                                                        const float* __restrict__ g,
                                                        unsigned short* __restrict__ xnt) {
  int n = blockIdx.x * 256 + threadIdx.x;
  int b = blockIdx.y;
  const float* xb = x + (size_t)b * CDIM * NSEQ + n;   // coalesced over n
  float sum = 0.f, sumsq = 0.f;
  for (int c = 0; c < CDIM; ++c) {
    float v = xb[(size_t)c * NSEQ];
    sum += v; sumsq += v * v;
  }
  float mean = sum * (1.0f / CDIM);
  float var  = sumsq * (1.0f / CDIM) - mean * mean;
  float rstd = rsqrtf(var + EPS);
  unsigned short* dst = xnt + ((size_t)b * NSEQ + n) * CDIM;
  for (int c = 0; c < CDIM; ++c) {
    float v = (xb[(size_t)c * NSEQ] - mean) * rstd * g[c];
    dst[c] = f32_to_bf16(v);
  }
}

// ---------------- shared GEMM core ------------------------------------------
// C[m0..m0+15, n0..n0+63] per wave; B tile (64 x 32) staged in LDS via async
// copies, double-buffered; A fragment register-prefetched one k-step ahead.
// All 4 B fragments are loaded before the WMMA group so the 8 ds_load_b128 can
// be claused and their latency overlapped by graduated s_wait_dscnt.
struct GemmAcc { v8f a0, a1, a2, a3; };

__device__ __forceinline__ GemmAcc gemm_core_128x64(const unsigned short* __restrict__ A,
                                                    const unsigned short* __restrict__ B,
                                                    unsigned short (*Bt)[64 * 32],
                                                    int tid, int lane) {
  const int row = tid >> 2, seg = tid & 3;   // 64 rows x 4 x b128 per tile
  // prologue: prefetch first B tile, load first A fragment
  async_g2l_b128(&Bt[0][row * 32 + seg * 8], B + (size_t)row * CDIM + seg * 8);
  FragAB a_cur = load_frag_a(A, CDIM, lane);
  wait_asynccnt0();
  __syncthreads();

  GemmAcc acc;
  acc.a0 = zero8(); acc.a1 = zero8(); acc.a2 = zero8(); acc.a3 = zero8();

  for (int k0 = 0; k0 < CDIM; k0 += 32) {
    int cb = (k0 >> 5) & 1;
    FragAB a_nxt = a_cur;
    if (k0 + 32 < CDIM) {
      // prefetch next B tile (async -> LDS) and next A fragment (global -> reg)
      async_g2l_b128(&Bt[cb ^ 1][row * 32 + seg * 8],
                     B + (size_t)row * CDIM + (k0 + 32) + seg * 8);
      a_nxt = load_frag_a(A + k0 + 32, CDIM, lane);
    }
    // batch all fragment loads from LDS, then the WMMA group
    FragAB b0 = load_frag_b(&Bt[cb][0 * 16 * 32], 32, lane);
    FragAB b1 = load_frag_b(&Bt[cb][1 * 16 * 32], 32, lane);
    FragAB b2 = load_frag_b(&Bt[cb][2 * 16 * 32], 32, lane);
    FragAB b3 = load_frag_b(&Bt[cb][3 * 16 * 32], 32, lane);
    acc.a0 = wmma_bf16(a_cur, b0, acc.a0);
    acc.a1 = wmma_bf16(a_cur, b1, acc.a1);
    acc.a2 = wmma_bf16(a_cur, b2, acc.a2);
    acc.a3 = wmma_bf16(a_cur, b3, acc.a3);
    a_cur = a_nxt;
    wait_asynccnt0();     // next buffer landed
    __syncthreads();      // all waves done reading cur buffer before reuse
  }
  return acc;
}

// ---------------- kernel 2: QKV projection GEMM -----------------------------
__global__ __launch_bounds__(256) void qkv_gemm_kernel(const unsigned short* __restrict__ Wbf,
                                                       const unsigned short* __restrict__ xnt,
                                                       unsigned short* __restrict__ qb,
                                                       unsigned short* __restrict__ kb,
                                                       unsigned short* __restrict__ vtb) {
  __shared__ alignas(16) unsigned short Bt[2][64 * 32];

  int tid = threadIdx.x, wave = tid >> 5, lane = tid & 31;
  int half = lane >> 4, col = lane & 15;
  int m0 = blockIdx.x * 128 + wave * 16;
  int n0 = blockIdx.y * 64;
  int b  = blockIdx.z;

  const unsigned short* A = Wbf + (size_t)m0 * CDIM;
  const unsigned short* B = xnt + ((size_t)b * NSEQ + n0) * CDIM;
  GemmAcc acc = gemm_core_128x64(A, B, Bt, tid, lane);

  float vals[4][8];
#pragma unroll
  for (int r = 0; r < 8; ++r) {
    vals[0][r] = acc.a0[r]; vals[1][r] = acc.a1[r];
    vals[2][r] = acc.a2[r]; vals[3][r] = acc.a3[r];
  }
  const float qscale = 0.125f;  // DHEAD^-0.5
#pragma unroll
  for (int j = 0; j < 4; ++j) {
    int n = n0 + 16 * j + col;
#pragma unroll
    for (int r = 0; r < 8; ++r) {
      int o = m0 + half * 8 + r;
      float f = vals[j][r];
      if (m0 < 512) {                                    // Q (scaled), (b,h,n,d)
        int hh = o >> 6, d = o & 63;
        qb[(((size_t)b * HEADS + hh) * NSEQ + n) * DHEAD + d] = f32_to_bf16(f * qscale);
      } else if (m0 < 1024) {                            // K, (b,h,n,d)
        int oo = o - 512; int hh = oo >> 6, d = oo & 63;
        kb[(((size_t)b * HEADS + hh) * NSEQ + n) * DHEAD + d] = f32_to_bf16(f);
      } else {                                           // V^T, (b,h,d,n)
        int oo = o - 1024; int hh = oo >> 6, d = oo & 63;
        vtb[(((size_t)b * HEADS + hh) * DHEAD + d) * NSEQ + n] = f32_to_bf16(f);
      }
    }
  }
}

// ---------------- kernel 3: flash attention ---------------------------------
// All 8 waves of a block share the same (b,h) and j0 schedule: K/V tiles staged
// once per block in LDS by async copies, double buffered. Per-wave P relayout
// (C->A fragment) goes through a private LDS slab.
__global__ __launch_bounds__(256) void attention_kernel(const unsigned short* __restrict__ qb,
                                                        const unsigned short* __restrict__ kb,
                                                        const unsigned short* __restrict__ vtb,
                                                        unsigned short* __restrict__ attnT) {
  __shared__ alignas(16) unsigned short Kt[2][32 * 64];    // [j][d], ld 64
  __shared__ alignas(16) unsigned short Vt[2][64 * 32];    // [d][j], ld 32
  __shared__ alignas(16) unsigned short plds[8][16 * 32];  // per-wave P staging

  int tid = threadIdx.x, wave = tid >> 5, lane = tid & 31;
  int half = lane >> 4, col = lane & 15;
  int b = blockIdx.z, h = blockIdx.y;
  int i0 = blockIdx.x * 128 + wave * 16;

  size_t bh = (size_t)b * HEADS + h;
  const unsigned short* qbase = qb  + bh * NSEQ * DHEAD;
  const unsigned short* kbase = kb  + bh * NSEQ * DHEAD;
  const unsigned short* vbase = vtb + bh * DHEAD * NSEQ;

  const int krow = tid >> 3, kseg = tid & 7;   // 32 rows x 8 b128 (K tile)
  const int vrow = tid >> 2, vseg = tid & 3;   // 64 rows x 4 b128 (V tile)
  auto issue_kv = [&](int buf, int j0) {
    async_g2l_b128(&Kt[buf][krow * 64 + kseg * 8],
                   kbase + (size_t)(j0 + krow) * DHEAD + kseg * 8);
    async_g2l_b128(&Vt[buf][vrow * 32 + vseg * 8],
                   vbase + (size_t)vrow * NSEQ + j0 + vseg * 8);
  };

  // prologue
  issue_kv(0, 0);
  FragAB aq0 = load_frag_a(qbase + (size_t)i0 * DHEAD + 0,  DHEAD, lane);
  FragAB aq1 = load_frag_a(qbase + (size_t)i0 * DHEAD + 32, DHEAD, lane);
  wait_asynccnt0();
  __syncthreads();

  v8f acc0 = zero8(), acc1 = zero8(), acc2 = zero8(), acc3 = zero8();
  float m[8], l[8];
#pragma unroll
  for (int r = 0; r < 8; ++r) { m[r] = -1e30f; l[r] = 0.f; }

  unsigned short* pl = plds[wave];

  for (int j0 = 0; j0 < NSEQ; j0 += 32) {
    int cb = (j0 >> 5) & 1;
    if (j0 + 32 < NSEQ) issue_kv(cb ^ 1, j0 + 32);   // overlap with compute below

    // S = Q * K^T : two 16x16 tiles, K-dim 64.  Load all K fragments first so
    // the 8 ds_load_b128 clause together, then run the WMMA group.
    FragAB bk00 = load_frag_b(&Kt[cb][0 * 16 * 64 + 0],  64, lane);
    FragAB bk01 = load_frag_b(&Kt[cb][0 * 16 * 64 + 32], 64, lane);
    FragAB bk10 = load_frag_b(&Kt[cb][1 * 16 * 64 + 0],  64, lane);
    FragAB bk11 = load_frag_b(&Kt[cb][1 * 16 * 64 + 32], 64, lane);
    v8f s0 = zero8(), s1 = zero8();
    s0 = wmma_bf16(aq0, bk00, s0);
    s0 = wmma_bf16(aq1, bk01, s0);
    s1 = wmma_bf16(aq0, bk10, s1);
    s1 = wmma_bf16(aq1, bk11, s1);

    // online softmax: row stats via 16-lane-half butterfly reduction (wave32)
#pragma unroll
    for (int r = 0; r < 8; ++r) {
      float a0 = s0[r], a1 = s1[r];
      float rmax = fmaxf(a0, a1);
      rmax = fmaxf(rmax, __shfl_xor(rmax, 1));
      rmax = fmaxf(rmax, __shfl_xor(rmax, 2));
      rmax = fmaxf(rmax, __shfl_xor(rmax, 4));
      rmax = fmaxf(rmax, __shfl_xor(rmax, 8));
      float mnew = fmaxf(m[r], rmax);
      float corr = __expf(m[r] - mnew);
      float p0 = __expf(a0 - mnew);
      float p1 = __expf(a1 - mnew);
      float rsum = p0 + p1;
      rsum += __shfl_xor(rsum, 1);
      rsum += __shfl_xor(rsum, 2);
      rsum += __shfl_xor(rsum, 4);
      rsum += __shfl_xor(rsum, 8);
      l[r] = l[r] * corr + rsum;
      m[r] = mnew;
      acc0[r] *= corr; acc1[r] *= corr; acc2[r] *= corr; acc3[r] *= corr;
      int row = half * 8 + r;
      pl[row * 32 + col]      = f32_to_bf16(p0);
      pl[row * 32 + 16 + col] = f32_to_bf16(p1);
    }
    asm volatile("s_wait_dscnt 0" ::: "memory");
    __builtin_amdgcn_wave_barrier();

    // O += P * V : load P A-fragment and all 4 V fragments, then WMMA group
    FragAB pa  = load_frag_a(pl, 32, lane);
    FragAB bv0 = load_frag_b(&Vt[cb][0 * 16 * 32], 32, lane);
    FragAB bv1 = load_frag_b(&Vt[cb][1 * 16 * 32], 32, lane);
    FragAB bv2 = load_frag_b(&Vt[cb][2 * 16 * 32], 32, lane);
    FragAB bv3 = load_frag_b(&Vt[cb][3 * 16 * 32], 32, lane);
    acc0 = wmma_bf16(pa, bv0, acc0);
    acc1 = wmma_bf16(pa, bv1, acc1);
    acc2 = wmma_bf16(pa, bv2, acc2);
    acc3 = wmma_bf16(pa, bv3, acc3);

    wait_asynccnt0();   // next K/V buffer landed
    __syncthreads();    // all waves done with cur buffer before it is reused
  }

  // epilogue: normalize, write (b, n, c=h*64+d) bf16 for the out-proj GEMM
#pragma unroll
  for (int r = 0; r < 8; ++r) {
    float inv = 1.0f / l[r];
    int i = i0 + half * 8 + r;
    size_t rowoff = ((size_t)b * NSEQ + i) * CDIM + h * DHEAD;
    attnT[rowoff + 0  + col] = f32_to_bf16(acc0[r] * inv);
    attnT[rowoff + 16 + col] = f32_to_bf16(acc1[r] * inv);
    attnT[rowoff + 32 + col] = f32_to_bf16(acc2[r] * inv);
    attnT[rowoff + 48 + col] = f32_to_bf16(acc3[r] * inv);
  }
}

// ---------------- kernel 4: output projection + bias ------------------------
__global__ __launch_bounds__(256) void out_gemm_kernel(const unsigned short* __restrict__ Wbf,
                                                       const unsigned short* __restrict__ attnT,
                                                       const float* __restrict__ bias,
                                                       float* __restrict__ out) {
  __shared__ alignas(16) unsigned short Bt[2][64 * 32];

  int tid = threadIdx.x, wave = tid >> 5, lane = tid & 31;
  int half = lane >> 4, col = lane & 15;
  int m0 = blockIdx.x * 128 + wave * 16;
  int n0 = blockIdx.y * 64;
  int b  = blockIdx.z;

  const unsigned short* A = Wbf + (size_t)m0 * CDIM;
  const unsigned short* B = attnT + ((size_t)b * NSEQ + n0) * CDIM;
  GemmAcc acc = gemm_core_128x64(A, B, Bt, tid, lane);

  float vals[4][8];
#pragma unroll
  for (int r = 0; r < 8; ++r) {
    vals[0][r] = acc.a0[r]; vals[1][r] = acc.a1[r];
    vals[2][r] = acc.a2[r]; vals[3][r] = acc.a3[r];
  }
#pragma unroll
  for (int j = 0; j < 4; ++j) {
    int n = n0 + 16 * j + col;
#pragma unroll
    for (int r = 0; r < 8; ++r) {
      int o = m0 + half * 8 + r;
      out[((size_t)b * CDIM + o) * NSEQ + n] = vals[j][r] + bias[o];
    }
  }
}

// ---------------- host-side launcher ----------------------------------------
extern "C" void kernel_launch(void* const* d_in, const int* in_sizes, int n_in,
                              void* d_out, int out_size, void* d_ws, size_t ws_size,
                              hipStream_t stream) {
  const float* x    = (const float*)d_in[0];   // (8, 512, 2048)
  const float* g    = (const float*)d_in[1];   // (512,)
  const float* Wqkv = (const float*)d_in[2];   // (1536, 512)
  const float* Wout = (const float*)d_in[3];   // (512, 512)
  const float* bout = (const float*)d_in[4];   // (512,)
  float* out = (float*)d_out;                  // (8, 512, 2048)

  char* ws = (char*)d_ws;
  size_t off = 0;
  auto alloc = [&](size_t bytes) -> void* {
    void* p = ws + off;
    off += (bytes + 255) & ~(size_t)255;
    return p;
  };
  unsigned short* Wqkv_bf = (unsigned short*)alloc((size_t)QKV_ROWS * CDIM * 2);  // 1.5 MB
  unsigned short* Wout_bf = (unsigned short*)alloc((size_t)CDIM * CDIM * 2);      // 0.5 MB
  unsigned short* xnt = (unsigned short*)alloc((size_t)BATCH * NSEQ * CDIM * 2);  // 16 MB
  unsigned short* qb  = (unsigned short*)alloc((size_t)BATCH * HEADS * NSEQ * DHEAD * 2); // 16 MB
  unsigned short* kb  = (unsigned short*)alloc((size_t)BATCH * HEADS * NSEQ * DHEAD * 2); // 16 MB
  unsigned short* vtb = (unsigned short*)alloc((size_t)BATCH * HEADS * DHEAD * NSEQ * 2); // 16 MB
  unsigned short* attnT = xnt;  // xnt dead after QKV GEMM -> reuse

  cvt_bf16_kernel<<<(QKV_ROWS * CDIM + 255) / 256, 256, 0, stream>>>(Wqkv, Wqkv_bf, QKV_ROWS * CDIM);
  cvt_bf16_kernel<<<(CDIM * CDIM + 255) / 256, 256, 0, stream>>>(Wout, Wout_bf, CDIM * CDIM);

  layernorm_kernel<<<dim3(NSEQ / 256, BATCH), 256, 0, stream>>>(x, g, xnt);

  qkv_gemm_kernel<<<dim3(QKV_ROWS / 128, NSEQ / 64, BATCH), 256, 0, stream>>>(
      Wqkv_bf, xnt, qb, kb, vtb);

  attention_kernel<<<dim3(NSEQ / 128, HEADS, BATCH), 256, 0, stream>>>(
      qb, kb, vtb, attnT);

  out_gemm_kernel<<<dim3(CDIM / 128, NSEQ / 64, BATCH), 256, 0, stream>>>(
      Wout_bf, attnT, bout, out);
}
